// JointNet_8804682957465
// MI455X (gfx1250) — matrix operand order repaired and checked by reading, compiled, not run on hardware
//
#include <hip/hip_runtime.h>

// ---------------- problem dims ----------------
static constexpr int Bdim = 4;
static constexpr int Tdim = 256;
static constexpr int Udim = 64;
static constexpr int Hdim = 640;   // = D_ENC = D_PRED = HDIM
static constexpr int Vdim = 1024;

// ---------------- vector types ----------------
typedef __attribute__((ext_vector_type(8)))  unsigned short ushort8;
typedef __attribute__((ext_vector_type(16))) unsigned short ushort16;
typedef __attribute__((ext_vector_type(16))) __bf16         v16bf;
typedef __attribute__((ext_vector_type(8)))  float          v8f;

union BfPack { ushort16 u; v16bf b; };

__device__ __forceinline__ unsigned short f2bf(float f) {
  unsigned int x = __float_as_uint(f);
  x += 0x7fffu + ((x >> 16) & 1u);   // round-to-nearest-even
  return (unsigned short)(x >> 16);
}

// ---------------- kernel 1/2: small f32 projections Y = X*W + b ----------------
__global__ void proj_kernel(const float* __restrict__ X, const float* __restrict__ W,
                            const float* __restrict__ bias, float* __restrict__ Y, int R) {
  int idx = blockIdx.x * blockDim.x + threadIdx.x;
  if (idx >= R * Hdim) return;
  int r = idx / Hdim;
  int h = idx - r * Hdim;
  const float* xr = X + (size_t)r * Hdim;
  float s = bias[h];
  for (int k = 0; k < Hdim; ++k) s += xr[k] * W[(size_t)k * Hdim + h];
  Y[idx] = s;
}

// ------------- kernel 3: W_fc (H x V, f32) -> Wt (V x H, bf16 col-major) -------------
__global__ void wt_kernel(const float* __restrict__ W_fc, unsigned short* __restrict__ Wt) {
  int idx = blockIdx.x * blockDim.x + threadIdx.x;
  if (idx >= Hdim * Vdim) return;
  int vcol = idx / Hdim;
  int h    = idx - vcol * Hdim;
  Wt[idx] = f2bf(W_fc[(size_t)h * Vdim + vcol]);
}

// ------------- kernel 4: fused tanh-joint + bf16 WMMA GEMM + log_softmax -------------
// 32 rows (2 M-tiles) x 1024 cols per 256-thread block; each B fragment feeds 2 WMMAs.
__global__ void __launch_bounds__(256)
joint_kernel(const float* __restrict__ enc_act, const float* __restrict__ pred_act,
             const unsigned short* __restrict__ Wt, const float* __restrict__ b_fc,
             float* __restrict__ out) {
  __shared__ __align__(16) unsigned short Alds[32 * Hdim];  // 40 KB bf16 A tile
  __shared__ float Red[8][16];                               // cross-wave reduce

  const int tid  = threadIdx.x;
  const int row0 = blockIdx.x << 5;            // 32 rows of (b,t,u) per block
  const int b    = row0 >> 14;                 // / (T*U)
  const int t    = (row0 >> 6) & (Tdim - 1);
  const int u0   = row0 & (Udim - 1);          // 0 or 32

  const float* encRow   = enc_act  + (size_t)(b * Tdim + t)  * Hdim;
  const float* predBase = pred_act + (size_t)(b * Udim + u0) * Hdim;

  // Phase 1: A[m][k] = bf16(tanh(enc[t][k] + pred[u0+m][k])), m = 0..31
  for (int idx = tid; idx < 32 * Hdim; idx += 256) {
    int m = idx / Hdim;
    int k = idx - m * Hdim;
    Alds[idx] = f2bf(tanhf(encRow[k] + predBase[m * Hdim + k]));
  }
  __syncthreads();

  const int wave = tid >> 5;
  const int lane = tid & 31;
  const int half = lane >> 4;
  const int mrow = lane & 15;

  v8f acc[2][8];
#pragma unroll
  for (int mt = 0; mt < 2; ++mt)
#pragma unroll
    for (int i = 0; i < 8; ++i) acc[mt][i] = (v8f){0.f, 0.f, 0.f, 0.f, 0.f, 0.f, 0.f, 0.f};

  // Phase 2: 20 k-steps of v_wmma_f32_16x16x32_bf16; 8 N-tiles/wave, 2 M-tiles reuse each B frag
  const unsigned short* aptr0 = &Alds[(mrow)      * Hdim + half * 8];
  const unsigned short* aptr1 = &Alds[(16 + mrow) * Hdim + half * 8];
  for (int kk = 0; kk < Hdim / 32; ++kk) {
    const int kb = kk * 32;
    // A fragments per ISA layout: lane(m, half) -> K runs [half*8, +8) and [16+half*8, +8)
    ushort8 a00 = *(const ushort8*)(aptr0 + kb);
    ushort8 a01 = *(const ushort8*)(aptr0 + kb + 16);
    ushort8 a10 = *(const ushort8*)(aptr1 + kb);
    ushort8 a11 = *(const ushort8*)(aptr1 + kb + 16);
    BfPack ua0, ua1;
    ua0.u = __builtin_shufflevector(a00, a01, 0, 1, 2, 3, 4, 5, 6, 7, 8, 9, 10, 11, 12, 13, 14, 15);
    ua1.u = __builtin_shufflevector(a10, a11, 0, 1, 2, 3, 4, 5, 6, 7, 8, 9, 10, 11, 12, 13, 14, 15);
#pragma unroll
    for (int i = 0; i < 8; ++i) {
      // B fragment: lane n=mrow, K runs [half*8, +8) and [16+half*8, +8) in Wt row
      const int col = ((wave << 3) + i) * 16 + mrow;
      const unsigned short* wrow = Wt + (size_t)col * Hdim + kb + half * 8;
      ushort8 b0 = *(const ushort8*)(wrow);
      ushort8 b1 = *(const ushort8*)(wrow + 16);
      BfPack ub;
      ub.u = __builtin_shufflevector(b0, b1, 0, 1, 2, 3, 4, 5, 6, 7, 8, 9, 10, 11, 12, 13, 14, 15);
      acc[0][i] = __builtin_amdgcn_wmma_f32_16x16x32_bf16(
          false, ua0.b, false, ub.b, (short)0, acc[0][i], false, false);
      acc[1][i] = __builtin_amdgcn_wmma_f32_16x16x32_bf16(
          false, ua1.b, false, ub.b, (short)0, acc[1][i], false, false);
    }
  }

  // Bias. C/D layout: VGPR v, lanes 0-15 -> (M=v, N=lane); lanes 16-31 -> (M=v+8, N=lane-16)
#pragma unroll
  for (int i = 0; i < 8; ++i) {
    float bias = b_fc[((wave << 3) + i) * 16 + mrow];
#pragma unroll
    for (int mt = 0; mt < 2; ++mt)
#pragma unroll
      for (int v = 0; v < 8; ++v) acc[mt][i][v] += bias;
  }

  // Log-softmax epilogue, one 16-row M-tile at a time.
#pragma unroll
  for (int mt = 0; mt < 2; ++mt) {
    // Row max: lane-local over 8 N-tiles, shuffle over the 16-lane half, cross-wave via LDS.
    float rmax[8];
#pragma unroll
    for (int v = 0; v < 8; ++v) {
      float x = acc[mt][0][v];
#pragma unroll
      for (int i = 1; i < 8; ++i) x = fmaxf(x, acc[mt][i][v]);
      rmax[v] = x;
    }
#pragma unroll
    for (int off = 8; off >= 1; off >>= 1)
#pragma unroll
      for (int v = 0; v < 8; ++v) rmax[v] = fmaxf(rmax[v], __shfl_xor(rmax[v], off, 32));

    if (mrow == 0) {  // lane 0 (rows 0-7) and lane 16 (rows 8-15)
#pragma unroll
      for (int v = 0; v < 8; ++v) Red[wave][half * 8 + v] = rmax[v];
    }
    __syncthreads();
    float gmax[8];
#pragma unroll
    for (int v = 0; v < 8; ++v) {
      float x = Red[0][half * 8 + v];
#pragma unroll
      for (int w = 1; w < 8; ++w) x = fmaxf(x, Red[w][half * 8 + v]);
      gmax[v] = x;
    }
    __syncthreads();

    // Sum of exp
    float rsum[8];
#pragma unroll
    for (int v = 0; v < 8; ++v) {
      float s = 0.f;
#pragma unroll
      for (int i = 0; i < 8; ++i) s += __expf(acc[mt][i][v] - gmax[v]);
      rsum[v] = s;
    }
#pragma unroll
    for (int off = 8; off >= 1; off >>= 1)
#pragma unroll
      for (int v = 0; v < 8; ++v) rsum[v] += __shfl_xor(rsum[v], off, 32);

    if (mrow == 0) {
#pragma unroll
      for (int v = 0; v < 8; ++v) Red[wave][half * 8 + v] = rsum[v];
    }
    __syncthreads();
    float lse[8];
#pragma unroll
    for (int v = 0; v < 8; ++v) {
      float s = 0.f;
#pragma unroll
      for (int w = 0; w < 8; ++w) s += Red[w][half * 8 + v];
      lse[v] = gmax[v] + __logf(s);
    }
    __syncthreads();  // Red reused by next mt iteration

    // Write log-softmax for this M-tile
    float* outRow = out + (size_t)(row0 + mt * 16) * Vdim;
#pragma unroll
    for (int i = 0; i < 8; ++i) {
      const int col = ((wave << 3) + i) * 16 + mrow;
#pragma unroll
      for (int v = 0; v < 8; ++v) {
        const int m = half * 8 + v;
        outRow[(size_t)m * Vdim + col] = acc[mt][i][v] - lse[v];
      }
    }
  }
}

// ---------------- launch ----------------
extern "C" void kernel_launch(void* const* d_in, const int* in_sizes, int n_in,
                              void* d_out, int out_size, void* d_ws, size_t ws_size,
                              hipStream_t stream) {
  const float* enc_out  = (const float*)d_in[0];
  const float* pred_out = (const float*)d_in[1];
  const float* W_enc    = (const float*)d_in[2];
  const float* b_enc    = (const float*)d_in[3];
  const float* W_pred   = (const float*)d_in[4];
  const float* b_pred   = (const float*)d_in[5];
  const float* W_fc     = (const float*)d_in[6];
  const float* b_fc     = (const float*)d_in[7];

  // workspace layout
  constexpr size_t ENC_BYTES  = (size_t)Bdim * Tdim * Hdim * sizeof(float);  // 2,621,440
  constexpr size_t PRED_BYTES = (size_t)Bdim * Udim * Hdim * sizeof(float);  //   655,360
  float*          enc_act  = (float*)d_ws;
  float*          pred_act = (float*)((char*)d_ws + ENC_BYTES);
  unsigned short* Wt       = (unsigned short*)((char*)d_ws + ENC_BYTES + PRED_BYTES);

  const int Renc  = Bdim * Tdim;  // 1024
  const int Rpred = Bdim * Udim;  // 256

  proj_kernel<<<(Renc * Hdim + 255) / 256, 256, 0, stream>>>(enc_out, W_enc, b_enc, enc_act, Renc);
  proj_kernel<<<(Rpred * Hdim + 255) / 256, 256, 0, stream>>>(pred_out, W_pred, b_pred, pred_act, Rpred);
  wt_kernel<<<(Hdim * Vdim + 255) / 256, 256, 0, stream>>>(W_fc, Wt);

  const int nTiles = (Bdim * Tdim * Udim) / 32;  // 2048
  joint_kernel<<<nTiles, 256, 0, stream>>>(enc_act, pred_act, Wt, b_fc, (float*)d_out);
}
